// _ObjectAttentionBlock_9337258902032
// MI455X (gfx1250) — matrix operand order repaired and checked by reading, not compiled, not run on hardware
//
#include <hip/hip_runtime.h>

#define EPSB 1e-5f
#define C_IN 512
#define KCH  256
#define HWSZ 16384
#define NK   19
#define TP   128
#define MSTR 264   // s_mat row stride in bf16 elems (256 + 8 pad, keeps 16B alignment)

typedef __attribute__((ext_vector_type(16))) __bf16 bfx16;
typedef __attribute__((ext_vector_type(8)))  __bf16 bfx8;
typedef __attribute__((ext_vector_type(4)))  __bf16 bfx4;
typedef __attribute__((ext_vector_type(8)))  float  fx8;
typedef __attribute__((ext_vector_type(4)))  float  f32x4;

// ---------------------------------------------------------------------------
// Fold eval-BN scale into weights, emit bf16 in WMMA A-fragment lane layout:
// fragment f = rb*(Cin/32)+kb holds a 16x32 tile; lane l stores 16 contiguous
// bf16: M = rb*16+(l&15), K = kb*32 + 16*hi + s (hi = l>>4, contiguous K).
// ---------------------------------------------------------------------------
__global__ __launch_bounds__(256) void k_prep_w(const float* __restrict__ w,
    const float* __restrict__ g, const float* __restrict__ b,
    const float* __restrict__ m, const float* __restrict__ v,
    int Cout, int Cin, __bf16* __restrict__ wf, float* __restrict__ bias)
{
  int e = blockIdx.x * 256 + threadIdx.x;
  int total = Cout * Cin;
  if (e < total) {
    int nkb = Cin >> 5;
    int f = e >> 9, l = (e >> 4) & 31, s = e & 15;
    int rb = f / nkb, kb = f - rb * nkb;
    int M = rb * 16 + (l & 15), hi = l >> 4;
    int vv = s >> 1, odd = s & 1;
    int K = kb * 32 + 16 * hi + s;
    float inv = g[M] * rsqrtf(v[M] + EPSB);
    wf[e] = (__bf16)(w[(size_t)M * Cin + K] * inv);
  }
  if (e < Cout) {
    float inv = g[e] * rsqrtf(v[e] + EPSB);
    bias[e] = b[e] - m[e] * inv;
  }
}

// ---------------------------------------------------------------------------
// k = BN∘ReLU(Wo2 · BN∘ReLU(Wo1 · proxy)), v = BN∘ReLU(Wd · proxy). Tiny.
// ---------------------------------------------------------------------------
__global__ __launch_bounds__(256) void k_prep_kv(const float* __restrict__ proxy,
    const float* __restrict__ w_o1, const float* __restrict__ o1g, const float* __restrict__ o1b,
    const float* __restrict__ o1m, const float* __restrict__ o1v,
    const float* __restrict__ w_o2, const float* __restrict__ o2g, const float* __restrict__ o2b,
    const float* __restrict__ o2m, const float* __restrict__ o2v,
    const float* __restrict__ w_d,  const float* __restrict__ dg,  const float* __restrict__ db,
    const float* __restrict__ dm,  const float* __restrict__ dv,
    float* __restrict__ kmat, float* __restrict__ vmat)
{
  __shared__ float t[KCH * NK];
  int n = blockIdx.x, o = threadIdx.x;
  const float* pn = proxy + (size_t)n * C_IN * NK;
  float acc[NK];
  #pragma unroll
  for (int j = 0; j < NK; ++j) acc[j] = 0.f;
  for (int c = 0; c < C_IN; ++c) {
    float wv = w_o1[o * C_IN + c];
    #pragma unroll
    for (int j = 0; j < NK; ++j) acc[j] = fmaf(wv, pn[c * NK + j], acc[j]);
  }
  {
    float inv = o1g[o] * rsqrtf(o1v[o] + EPSB);
    float bb = o1b[o] - o1m[o] * inv;
    #pragma unroll
    for (int j = 0; j < NK; ++j) t[o * NK + j] = fmaxf(acc[j] * inv + bb, 0.f);
  }
  __syncthreads();
  #pragma unroll
  for (int j = 0; j < NK; ++j) acc[j] = 0.f;
  for (int c = 0; c < KCH; ++c) {
    float wv = w_o2[o * KCH + c];
    #pragma unroll
    for (int j = 0; j < NK; ++j) acc[j] = fmaf(wv, t[c * NK + j], acc[j]);
  }
  {
    float inv = o2g[o] * rsqrtf(o2v[o] + EPSB);
    float bb = o2b[o] - o2m[o] * inv;
    #pragma unroll
    for (int j = 0; j < NK; ++j)
      kmat[((size_t)n * KCH + o) * NK + j] = fmaxf(acc[j] * inv + bb, 0.f);
  }
  #pragma unroll
  for (int j = 0; j < NK; ++j) acc[j] = 0.f;
  for (int c = 0; c < C_IN; ++c) {
    float wv = w_d[o * C_IN + c];
    #pragma unroll
    for (int j = 0; j < NK; ++j) acc[j] = fmaf(wv, pn[c * NK + j], acc[j]);
  }
  {
    float inv = dg[o] * rsqrtf(dv[o] + EPSB);
    float bb = db[o] - dm[o] * inv;
    #pragma unroll
    for (int j = 0; j < NK; ++j)
      vmat[((size_t)n * KCH + o) * NK + j] = fmaxf(acc[j] * inv + bb, 0.f);
  }
}

// ---------------------------------------------------------------------------
// Fused: GEMM1 -> GEMM2 -> softmax attention -> GEMM3, one 128-pixel tile per
// workgroup (8 waves). s_mat time-shares t1 / q / ctx (barrier-separated).
// x slab staging is register-pipelined + LDS double-buffered so the next
// slab's global (NT) loads fly under the current slab's WMMAs.
// ---------------------------------------------------------------------------
__global__ __launch_bounds__(256) void k_fused(
    const float* __restrict__ x,
    const __bf16* __restrict__ w1f, const float* __restrict__ b1,
    const __bf16* __restrict__ w2f, const float* __restrict__ b2,
    const __bf16* __restrict__ wuf, const float* __restrict__ bu,
    const float* __restrict__ kmat, const float* __restrict__ vmat,
    float* __restrict__ out)
{
  __shared__ __bf16 s_mat[TP * MSTR];      // 67.6 KB: t1 -> q -> ctx  [pixel][channel]
  __shared__ __bf16 s_xf[2][8 * 32 * 16];  // 16 KB: x slab double buffer, B-frag layout
  __shared__ float  s_sim[TP * 20];        // 10 KB: softmaxed attention rows

  const int tid  = threadIdx.x;
  const int wv   = tid >> 5;
  const int lane = tid & 31;
  const int n    = blockIdx.x >> 7;
  const int p0   = (blockIdx.x & 127) * TP;
  const float* xn = x + (size_t)n * C_IN * HWSZ + p0;
  const int px = lane & 15;
  const int hi = lane >> 4;

  // slab staging geometry: thread handles 4 channels x 4 pixels
  const int p4   = (tid & 31) << 2;        // pixel group base (0..124)
  const int cl0  = (tid >> 5) << 2;        // channel-in-slab base (0..28)
  const int hic  = cl0 >> 4;               // which 16-channel half
  const int slt  = cl0 & 15;               // slot base within lane entry

  fx8 zero = {};
  fx8 acc[2][8];
  #pragma unroll
  for (int i = 0; i < 2; ++i)
    #pragma unroll
    for (int j = 0; j < 8; ++j) acc[i][j] = zero;

  f32x4 xr[4];
  #pragma unroll
  for (int i = 0; i < 4; ++i)   // preload slab 0 (nontemporal: streamed once)
    xr[i] = __builtin_nontemporal_load(
        (const f32x4*)(xn + (size_t)(cl0 + i) * HWSZ + p4));

  // ---------------- GEMM1: t1 = ReLU(BN(W1 * Xtile))  [256 x 128] -----------
  for (int kb = 0; kb < 16; ++kb) {
    const int buf = kb & 1;
    // convert + store current slab regs -> LDS (packed 4-channel b64 stores)
    #pragma unroll
    for (int j = 0; j < 4; ++j) {
      bfx4 pk;
      #pragma unroll
      for (int i = 0; i < 4; ++i) pk[i] = (__bf16)xr[i][j];
      int p = p4 + j;
      int entry = (((p >> 4) << 5) | (p & 15) | (hic << 4));
      *(bfx4*)(s_xf[buf] + (entry << 4) + slt) = pk;
    }
    // issue next slab's global loads now; they complete under the WMMAs below
    if (kb < 15) {
      #pragma unroll
      for (int i = 0; i < 4; ++i)
        xr[i] = __builtin_nontemporal_load(
            (const f32x4*)(xn + (size_t)(((kb + 1) << 5) + cl0 + i) * HWSZ + p4));
    }
    if (kb < 14)  // warm L2 two slabs ahead -> global_prefetch_b8
      __builtin_prefetch(xn + (size_t)(((kb + 2) << 5) + (tid & 31)) * HWSZ + ((tid >> 5) << 4), 0, 1);
    __syncthreads();
    bfx16 a0 = *(const bfx16*)(w1f + ((((2 * wv)     * 16 + kb) << 9) + (lane << 4)));
    bfx16 a1 = *(const bfx16*)(w1f + ((((2 * wv + 1) * 16 + kb) << 9) + (lane << 4)));
    #pragma unroll
    for (int cb = 0; cb < 8; ++cb) {
      bfx16 bm = *(const bfx16*)(s_xf[buf] + (((cb << 5) | lane) << 4));
      acc[0][cb] = __builtin_amdgcn_wmma_f32_16x16x32_bf16(false, a0, false, bm, (short)0, acc[0][cb], false, false);
      acc[1][cb] = __builtin_amdgcn_wmma_f32_16x16x32_bf16(false, a1, false, bm, (short)0, acc[1][cb], false, false);
    }
    __syncthreads();   // all waves done with s_xf[buf] before it is refilled
  }
  #pragma unroll
  for (int rbi = 0; rbi < 2; ++rbi) {
    int rowb = (2 * wv + rbi) * 16 + hi * 8;
    #pragma unroll
    for (int cb = 0; cb < 8; ++cb) {
      int p = (cb << 4) | px;
      bfx8 pk;
      #pragma unroll
      for (int e = 0; e < 8; ++e)
        pk[e] = (__bf16)fmaxf(acc[rbi][cb][e] + b1[rowb + e], 0.f);
      *(bfx8*)(s_mat + p * MSTR + rowb) = pk;
    }
  }
  __syncthreads();

  // ---------------- GEMM2: q = ReLU(BN(W2 * t1))  [256 x 128] --------------
  #pragma unroll
  for (int i = 0; i < 2; ++i)
    #pragma unroll
    for (int j = 0; j < 8; ++j) acc[i][j] = zero;
  for (int kb = 0; kb < 8; ++kb) {
    bfx16 a0 = *(const bfx16*)(w2f + ((((2 * wv)     * 8 + kb) << 9) + (lane << 4)));
    bfx16 a1 = *(const bfx16*)(w2f + ((((2 * wv + 1) * 8 + kb) << 9) + (lane << 4)));
    int chunk = (kb << 5) + (hi << 4);
    #pragma unroll
    for (int cb = 0; cb < 8; ++cb) {
      bfx16 bm = *(const bfx16*)(s_mat + ((cb << 4) | px) * MSTR + chunk);
      acc[0][cb] = __builtin_amdgcn_wmma_f32_16x16x32_bf16(false, a0, false, bm, (short)0, acc[0][cb], false, false);
      acc[1][cb] = __builtin_amdgcn_wmma_f32_16x16x32_bf16(false, a1, false, bm, (short)0, acc[1][cb], false, false);
    }
  }
  __syncthreads();   // all t1 reads done -> safe to overwrite s_mat with q
  #pragma unroll
  for (int rbi = 0; rbi < 2; ++rbi) {
    int rowb = (2 * wv + rbi) * 16 + hi * 8;
    #pragma unroll
    for (int cb = 0; cb < 8; ++cb) {
      int p = (cb << 4) | px;
      bfx8 pk;
      #pragma unroll
      for (int e = 0; e < 8; ++e)
        pk[e] = (__bf16)fmaxf(acc[rbi][cb][e] + b2[rowb + e], 0.f);
      *(bfx8*)(s_mat + p * MSTR + rowb) = pk;
    }
  }
  __syncthreads();

  // ---------------- Attention: sim = softmax(qT k / 16)  [128 x 19] --------
  const float* kn = kmat + (size_t)n * KCH * NK;
  const float* vn = vmat + (size_t)n * KCH * NK;
  if (tid < TP) {
    int p = tid;
    float sim[NK];
    #pragma unroll
    for (int j = 0; j < NK; ++j) sim[j] = 0.f;
    const __bf16* qr = s_mat + p * MSTR;
    for (int c8 = 0; c8 < KCH; c8 += 8) {
      bfx8 q8 = *(const bfx8*)(qr + c8);      // one ds_load_b128 per 8 channels
      #pragma unroll
      for (int u = 0; u < 8; ++u) {
        float qv = (float)q8[u];
        const float* kc = kn + (c8 + u) * NK; // wave-uniform address -> s_load
        #pragma unroll
        for (int j = 0; j < NK; ++j) sim[j] = fmaf(qv, kc[j], sim[j]);
      }
    }
    float mx = -1e30f;
    #pragma unroll
    for (int j = 0; j < NK; ++j) { sim[j] *= 0.0625f; mx = fmaxf(mx, sim[j]); }
    float sum = 0.f;
    #pragma unroll
    for (int j = 0; j < NK; ++j) { float e = __expf(sim[j] - mx); sim[j] = e; sum += e; }
    float rs = 1.f / sum;
    #pragma unroll
    for (int j = 0; j < NK; ++j) s_sim[p * 20 + j] = sim[j] * rs;
  }
  __syncthreads();   // q dead -> ctx may overwrite s_mat

  // ---------------- ctx = sim * vT  [128 x 256] -----------------------------
  {
    int p = tid & 127;
    int c0 = (tid >> 7) * 128;
    float sv[NK];
    #pragma unroll
    for (int j = 0; j < NK; ++j) sv[j] = s_sim[p * 20 + j];
    for (int c = c0; c < c0 + 128; c += 4) {
      bfx4 pk;
      #pragma unroll
      for (int u = 0; u < 4; ++u) {
        const float* vc = vn + (c + u) * NK;  // wave-uniform address -> s_load
        float a = 0.f;
        #pragma unroll
        for (int j = 0; j < NK; ++j) a = fmaf(sv[j], vc[j], a);
        pk[u] = (__bf16)a;
      }
      *(bfx4*)(s_mat + p * MSTR + c) = pk;    // ds_store_b64 per 4 channels
    }
  }
  __syncthreads();

  // ---------------- GEMM3: out = ReLU(BN(Wu * ctx))  [512 x 128] ------------
  const size_t ob = (size_t)n * C_IN * HWSZ + p0;
  for (int rp = 0; rp < 2; ++rp) {
    #pragma unroll
    for (int i = 0; i < 2; ++i)
      #pragma unroll
      for (int j = 0; j < 8; ++j) acc[i][j] = zero;
    for (int kb = 0; kb < 8; ++kb) {
      int rbA = rp * 16 + 2 * wv;
      bfx16 a0 = *(const bfx16*)(wuf + (((rbA    * 8 + kb) << 9) + (lane << 4)));
      bfx16 a1 = *(const bfx16*)(wuf + ((((rbA+1)* 8 + kb) << 9) + (lane << 4)));
      int chunk = (kb << 5) + (hi << 4);
      #pragma unroll
      for (int cb = 0; cb < 8; ++cb) {
        bfx16 bm = *(const bfx16*)(s_mat + ((cb << 4) | px) * MSTR + chunk);
        acc[0][cb] = __builtin_amdgcn_wmma_f32_16x16x32_bf16(false, a0, false, bm, (short)0, acc[0][cb], false, false);
        acc[1][cb] = __builtin_amdgcn_wmma_f32_16x16x32_bf16(false, a1, false, bm, (short)0, acc[1][cb], false, false);
      }
    }
    #pragma unroll
    for (int rbi = 0; rbi < 2; ++rbi) {
      int rowb = (rp * 16 + 2 * wv + rbi) * 16 + hi * 8;
      #pragma unroll
      for (int cb = 0; cb < 8; ++cb) {
        int p = (cb << 4) | px;
        #pragma unroll
        for (int e = 0; e < 8; ++e) {
          float o = fmaxf(acc[rbi][cb][e] + bu[rowb + e], 0.f);
          // streamed once: nontemporal store keeps L2 for weights/k/v
          __builtin_nontemporal_store(o, out + ob + (size_t)(rowb + e) * HWSZ + p);
        }
      }
    }
  }
}

// ---------------------------------------------------------------------------
extern "C" void kernel_launch(void* const* d_in, const int* in_sizes, int n_in,
                              void* d_out, int out_size, void* d_ws, size_t ws_size,
                              hipStream_t stream)
{
  (void)in_sizes; (void)n_in; (void)out_size; (void)ws_size;
  const float* x     = (const float*)d_in[0];
  const float* proxy = (const float*)d_in[1];
  const float* w_p1  = (const float*)d_in[2];
  const float* w_p2  = (const float*)d_in[3];
  const float* w_o1  = (const float*)d_in[4];
  const float* w_o2  = (const float*)d_in[5];
  const float* w_d   = (const float*)d_in[6];
  const float* w_u   = (const float*)d_in[7];
  const float* p1g = (const float*)d_in[8],  *p1b = (const float*)d_in[9];
  const float* p1m = (const float*)d_in[10], *p1v = (const float*)d_in[11];
  const float* p2g = (const float*)d_in[12], *p2b = (const float*)d_in[13];
  const float* p2m = (const float*)d_in[14], *p2v = (const float*)d_in[15];
  const float* o1g = (const float*)d_in[16], *o1b = (const float*)d_in[17];
  const float* o1m = (const float*)d_in[18], *o1v = (const float*)d_in[19];
  const float* o2g = (const float*)d_in[20], *o2b = (const float*)d_in[21];
  const float* o2m = (const float*)d_in[22], *o2v = (const float*)d_in[23];
  const float* dg  = (const float*)d_in[24], *db  = (const float*)d_in[25];
  const float* dm  = (const float*)d_in[26], *dv  = (const float*)d_in[27];
  const float* ug  = (const float*)d_in[28], *ub  = (const float*)d_in[29];
  const float* um  = (const float*)d_in[30], *uv  = (const float*)d_in[31];

  char* ws = (char*)d_ws;
  __bf16* w1f = (__bf16*)(ws);              // 256x512 bf16 = 262144 B
  __bf16* w2f = (__bf16*)(ws + 262144);     // 256x256 bf16 = 131072 B
  __bf16* wuf = (__bf16*)(ws + 393216);     // 512x256 bf16 = 262144 B
  float*  b1  = (float*)(ws + 655360);      // 1 KB
  float*  b2  = (float*)(ws + 656384);      // 1 KB
  float*  bu  = (float*)(ws + 657408);      // 2 KB
  float*  km  = (float*)(ws + 659456);      // 8*256*19 f32 = 155648 B
  float*  vm  = (float*)(ws + 815104);      // 155648 B

  k_prep_w<<<512, 256, 0, stream>>>(w_p1, p1g, p1b, p1m, p1v, 256, 512, w1f, b1);
  k_prep_w<<<256, 256, 0, stream>>>(w_p2, p2g, p2b, p2m, p2v, 256, 256, w2f, b2);
  k_prep_w<<<512, 256, 0, stream>>>(w_u,  ug,  ub,  um,  uv,  512, 256, wuf, bu);
  k_prep_kv<<<8, 256, 0, stream>>>(proxy,
      w_o1, o1g, o1b, o1m, o1v,
      w_o2, o2g, o2b, o2m, o2v,
      w_d,  dg,  db,  dm,  dv,
      km, vm);
  k_fused<<<1024, 256, 0, stream>>>(x, w1f, b1, w2f, b2, wuf, bu, km, vm, (float*)d_out);
}
